// HybridQLSTM_65481071400062
// MI455X (gfx1250) — matrix operand order repaired and checked by reading, not compiled, not run on hardware
//
#include <hip/hip_runtime.h>

// HybridQLSTM on gfx1250 (MI455X, wave32).
//
// Phase A (qlstm_xproj): time-parallel input projection for all 4 gates via
//   v_wmma_f32_16x16x32_f16. M = T*B = 1,048,576 rows, N = 64, K = 128.
//   Weights (64x128) converted to f16 fragments once per wave and held in
//   VGPRs; A tiles streamed from HBM (bandwidth-bound: ~0.75 GB traffic).
//   Bias folded into the store. Result Zx lives in d_ws (256 MiB f32).
//
// Phase B (qlstm_recur): sequential scan over T. lane = row2*16 + k, one wave
//   handles 2 batch rows. Recurrent h @ Wh.T via 16 shfl-broadcast + FMA,
//   cumprod(cos(z+theta)) as a 4-step log-scan across the 16-lane group,
//   fast sigmoid/tanh via v_exp/v_rcp.

typedef __attribute__((ext_vector_type(16))) _Float16 v16h;
typedef __attribute__((ext_vector_type(8)))  float    v8f;

#define T_LEN  512
#define BATCH  2048
#define DIN    128
#define HID    16
#define FAN    (DIN + HID)          // 144
#define NROW   (T_LEN * BATCH)      // 1048576 rows of the big GEMM
#define NTILE  (NROW / 16)          // 65536 M-tiles
#define NGCOL  64                   // 4 gates * 16

__device__ __forceinline__ float fast_sigmoid(float x) {
    return __builtin_amdgcn_rcpf(1.0f + __expf(-x));
}
__device__ __forceinline__ float fast_tanh(float x) {
    // 1 - 2/(exp(2x)+1); inputs here are bounded (|x| <= ~16), safe.
    return 1.0f - 2.0f * __builtin_amdgcn_rcpf(__expf(2.0f * x) + 1.0f);
}

// ---------------------------------------------------------------------------
// Phase A: Zx[t*B+b][g*16+n] = x_t[b] . Wg[n][0:128] + bg[n]
// ---------------------------------------------------------------------------
__global__ __launch_bounds__(256)
void qlstm_xproj(const float* __restrict__ X,
                 const float* __restrict__ Wf, const float* __restrict__ bf,
                 const float* __restrict__ Wi, const float* __restrict__ bi,
                 const float* __restrict__ Wu, const float* __restrict__ bu,
                 const float* __restrict__ Wo, const float* __restrict__ bo,
                 float* __restrict__ Z)
{
    const int lane = threadIdx.x & 31;
    const int wv   = threadIdx.x >> 5;
    const int half = lane >> 4;       // 0: lanes 0-15, 1: lanes 16-31
    const int mn   = lane & 15;       // A: row-in-tile, B: column (gate out)
    const int gwave  = blockIdx.x * 8 + wv;
    const int nwaves = gridDim.x * 8;

    const float* Wg[4] = {Wf, Wi, Wu, Wo};
    const float bias[4] = { bf[mn], bi[mn], bu[mn], bo[mn] };

    // Preload B fragments (K x N = 32 x 16), one per (k-chunk, gate).
    // B order used here (contiguous k, 32x16): lanes 0-15 hold K=0..15, lanes 16-31 K=16..31;
    // VGPR v holds K = 2v, 2v+1 -> half index j maps to K = half*16 + j.
    // B[k][n] = W[n][k] (row n of W is contiguous in k).
    v16h bfrag[4][4];
    #pragma unroll
    for (int kc = 0; kc < 4; ++kc) {
        #pragma unroll
        for (int g = 0; g < 4; ++g) {
            const float4* wp =
                (const float4*)(Wg[g] + mn * FAN + kc * 32 + half * 16);
            float4 q0 = wp[0], q1 = wp[1], q2 = wp[2], q3 = wp[3];
            v16h bv;
            bv[0]=(_Float16)q0.x; bv[1]=(_Float16)q0.y; bv[2]=(_Float16)q0.z; bv[3]=(_Float16)q0.w;
            bv[4]=(_Float16)q1.x; bv[5]=(_Float16)q1.y; bv[6]=(_Float16)q1.z; bv[7]=(_Float16)q1.w;
            bv[8]=(_Float16)q2.x; bv[9]=(_Float16)q2.y; bv[10]=(_Float16)q2.z; bv[11]=(_Float16)q2.w;
            bv[12]=(_Float16)q3.x; bv[13]=(_Float16)q3.y; bv[14]=(_Float16)q3.z; bv[15]=(_Float16)q3.w;
            bfrag[kc][g] = bv;
        }
    }

    for (int tile = gwave; tile < NTILE; tile += nwaves) {
        v8f acc[4] = {};
        const float* rowp = X + (size_t)(tile * 16 + mn) * DIN;
        #pragma unroll
        for (int kc = 0; kc < 4; ++kc) {
            // A order used here (contiguous k, 16x32): lane holds row mn; kb = half*16;
            // half idx j -> K = kb+j.
            const float4* ap = (const float4*)(rowp + kc * 32 + half * 16);
            float4 q0 = ap[0];   // K = kb .. kb+3
            float4 q1 = ap[1];   // K = kb+4 .. kb+7
            float4 q2 = ap[2];   // K = kb+8 .. kb+11
            float4 q3 = ap[3];   // K = kb+12 .. kb+15
            v16h av;
            av[0]=(_Float16)q0.x; av[1]=(_Float16)q0.y; av[2]=(_Float16)q0.z; av[3]=(_Float16)q0.w;
            av[4]=(_Float16)q1.x; av[5]=(_Float16)q1.y; av[6]=(_Float16)q1.z; av[7]=(_Float16)q1.w;
            av[8]=(_Float16)q2.x; av[9]=(_Float16)q2.y; av[10]=(_Float16)q2.z; av[11]=(_Float16)q2.w;
            av[12]=(_Float16)q3.x; av[13]=(_Float16)q3.y; av[14]=(_Float16)q3.z; av[15]=(_Float16)q3.w;
            #pragma unroll
            for (int g = 0; g < 4; ++g) {
                acc[g] = __builtin_amdgcn_wmma_f32_16x16x32_f16(
                    false, av, false, bfrag[kc][g], (short)0, acc[g],
                    false, false);
            }
        }
        // D layout (f32 16x16): lane holds column n = mn, VGPR v holds row
        // half*8 + v. Store Zx[row][g*16+n] with bias folded in.
        const size_t rbase = (size_t)tile * 16 + half * 8;
        #pragma unroll
        for (int g = 0; g < 4; ++g) {
            #pragma unroll
            for (int v = 0; v < 8; ++v) {
                Z[(rbase + v) * NGCOL + g * 16 + mn] = acc[g][v] + bias[g];
            }
        }
    }
}

// ---------------------------------------------------------------------------
// Phase B: sequential LSTM scan. One wave = 2 batch rows (lane = r*16 + k).
// ---------------------------------------------------------------------------
__global__ __launch_bounds__(256)
void qlstm_recur(const float* __restrict__ Z,
                 const float* __restrict__ Wf, const float* __restrict__ Wi,
                 const float* __restrict__ Wu, const float* __restrict__ Wo,
                 const float* __restrict__ thf, const float* __restrict__ thi,
                 const float* __restrict__ thu, const float* __restrict__ tho,
                 float* __restrict__ out)
{
    const int lane  = threadIdx.x & 31;
    const int wv    = threadIdx.x >> 5;
    const int k     = lane & 15;          // hidden index
    const int rbase = lane & 16;          // lane base of this row's 16-group
    const int b     = blockIdx.x * 16 + wv * 2 + (lane >> 4);

    // Recurrent weights: lane k holds Wh_g[k][0..15] (cols 128..143 of W).
    float whf[16], whi[16], whu[16], who[16];
    #pragma unroll
    for (int j = 0; j < 16; ++j) {
        whf[j] = Wf[k * FAN + DIN + j];
        whi[j] = Wi[k * FAN + DIN + j];
        whu[j] = Wu[k * FAN + DIN + j];
        who[j] = Wo[k * FAN + DIN + j];
    }
    const float tf = thf[k], ti = thi[k], tu = thu[k], to = tho[k];

    float h = 0.0f, c = 0.0f;
    float* hx_out = out + (size_t)T_LEN * BATCH * HID;
    float* cx_out = hx_out + (size_t)BATCH * HID;

    for (int t = 0; t < T_LEN; ++t) {
        const float* zr = Z + ((size_t)t * BATCH + b) * NGCOL;
        float zf = zr[k], zi = zr[16 + k], zu = zr[32 + k], zo = zr[48 + k];

        // z_g += sum_j h_j * Wh_g[k][j]  (broadcast h_j across the row group)
        #pragma unroll
        for (int j = 0; j < 16; ++j) {
            float hj = __shfl(h, rbase + j, 32);
            zf = fmaf(hj, whf[j], zf);
            zi = fmaf(hj, whi[j], zi);
            zu = fmaf(hj, whu[j], zu);
            zo = fmaf(hj, who[j], zo);
        }

        float pf = __cosf(zf + tf);
        float pi = __cosf(zi + ti);
        float pu = __cosf(zu + tu);
        float po = __cosf(zo + to);

        // Inclusive cumprod over k in 0..15 (log-scan within 16-lane group).
        #pragma unroll
        for (int d = 1; d < 16; d <<= 1) {
            float uf = __shfl(pf, (lane - d) & 31, 32);
            float ui = __shfl(pi, (lane - d) & 31, 32);
            float uu = __shfl(pu, (lane - d) & 31, 32);
            float uo = __shfl(po, (lane - d) & 31, 32);
            if (k >= d) { pf *= uf; pi *= ui; pu *= uu; po *= uo; }
        }

        float fg = fast_sigmoid(pf);
        float ig = fast_sigmoid(pi);
        float gg = fast_tanh(pu);
        float og = fast_sigmoid(po);

        c = fmaf(fg, c, ig * gg);
        h = og * fast_tanh(c);

        out[((size_t)t * BATCH + b) * HID + k] = h;
    }
    hx_out[(size_t)b * HID + k] = h;
    cx_out[(size_t)b * HID + k] = c;
}

// ---------------------------------------------------------------------------
extern "C" void kernel_launch(void* const* d_in, const int* in_sizes, int n_in,
                              void* d_out, int out_size, void* d_ws, size_t ws_size,
                              hipStream_t stream)
{
    const float* X   = (const float*)d_in[0];
    const float* Wf  = (const float*)d_in[1];
    const float* bf  = (const float*)d_in[2];
    const float* Wi  = (const float*)d_in[3];
    const float* bi  = (const float*)d_in[4];
    const float* Wu  = (const float*)d_in[5];
    const float* bu  = (const float*)d_in[6];
    const float* Wo  = (const float*)d_in[7];
    const float* bo  = (const float*)d_in[8];
    const float* thf = (const float*)d_in[9];
    const float* thi = (const float*)d_in[10];
    const float* thu = (const float*)d_in[11];
    const float* tho = (const float*)d_in[12];

    float* Z   = (float*)d_ws;   // needs NROW*64*4 = 256 MiB
    float* out = (float*)d_out;

    // Phase A: 1024 blocks x 8 waves = 8192 waves, 8 tiles each.
    qlstm_xproj<<<1024, 256, 0, stream>>>(X, Wf, bf, Wi, bi, Wu, bu, Wo, bo, Z);

    // Phase B: 128 blocks x 256 threads = B*H lanes.
    qlstm_recur<<<BATCH / 16, 256, 0, stream>>>(Z, Wf, Wi, Wu, Wo,
                                                thf, thi, thu, tho, out);
}
